// SimpleRNN_11802570130200
// MI455X (gfx1250) — matrix operand order repaired and checked by reading, not compiled, not run on hardware
//
#include <hip/hip_runtime.h>
#include <math.h>

typedef __attribute__((ext_vector_type(16))) __bf16 v16bf;
typedef __attribute__((ext_vector_type(8)))  __bf16 v8bf;
typedef __attribute__((ext_vector_type(8)))  float  v8f;

#define B_  64
#define T_  512
#define D_  256
#define U_  512

// ws layout (bytes):
//   [0,        262144)  Wpk : bf16 WMMA-B layout of W  [256x512]
//   [262144,   786432)  Upk : bf16 WMMA-B layout of U  [512x512]
//   [786432,   851968)  state buf 0 : bf16 row-major [64x512]
//   [851968,   917504)  state buf 1 : bf16 row-major [64x512]

// ---------------------------------------------------------------------------
// Pack fp32 weights [K,512] into bf16 WMMA B-fragment layout: for K-step s
// (32 K values) and 16-col block j, lane = (n%16) + 16*((k%32/8)&1) owns 16
// contiguous bf16 (element = k%8 + 8*(k%32>=16)) -> one aligned 32B load per WMMA.
// ---------------------------------------------------------------------------
__global__ void pack_weights(const float* __restrict__ W,
                             const float* __restrict__ Um,
                             __bf16* __restrict__ Wpk,
                             __bf16* __restrict__ Upk) {
  int idx = blockIdx.x * blockDim.x + threadIdx.x;
  const int nW = D_ * U_;
  const int total = nW + U_ * U_;
  if (idx >= total) return;
  const float* src;
  __bf16* dst;
  int K, k, n;
  if (idx < nW) { src = W;  dst = Wpk; K = D_; k = idx / U_;      n = idx % U_; }
  else          { src = Um; dst = Upk; K = U_; k = (idx-nW)/U_;   n = (idx-nW)%U_; }
  const int S    = K >> 5;
  const int j    = n >> 4;
  const int nl   = n & 15;
  const int s    = k >> 5;
  const int kr   = k & 31;
  const int lane = nl + ((kr & 8) ? 16 : 0);
  const int elem = (kr & 7) + ((kr & 16) ? 8 : 0);
  dst[((((size_t)j * S + s) * 32 + lane) << 4) + elem] = (__bf16)src[(size_t)k * U_ + n];
}

// Build bf16 A-fragment from fp32: elems 0..7 = K[k0..k0+7], 8..15 = K[k0+16..k0+23].
__device__ __forceinline__ v16bf make_afrag_f32(const float* __restrict__ p) {
  const float4 f0 = ((const float4*)p)[0];
  const float4 f1 = ((const float4*)p)[1];
  const float4 f2 = ((const float4*)p)[4];   // +16 floats
  const float4 f3 = ((const float4*)p)[5];   // +20 floats
  v16bf a;
  a[0]=(__bf16)f0.x;  a[1]=(__bf16)f0.y;  a[2]=(__bf16)f0.z;  a[3]=(__bf16)f0.w;
  a[4]=(__bf16)f1.x;  a[5]=(__bf16)f1.y;  a[6]=(__bf16)f1.z;  a[7]=(__bf16)f1.w;
  a[8]=(__bf16)f2.x;  a[9]=(__bf16)f2.y;  a[10]=(__bf16)f2.z; a[11]=(__bf16)f2.w;
  a[12]=(__bf16)f3.x; a[13]=(__bf16)f3.y; a[14]=(__bf16)f3.z; a[15]=(__bf16)f3.w;
  return a;
}

// Same fragment from a row-major bf16 state row: two aligned 16B loads.
__device__ __forceinline__ v16bf make_afrag_bf16(const __bf16* __restrict__ p) {
  v8bf lo = *(const v8bf*)(p);
  v8bf hi = *(const v8bf*)(p + 16);
  return __builtin_shufflevector(lo, hi, 0,1,2,3,4,5,6,7,8,9,10,11,12,13,14,15);
}

// ---------------------------------------------------------------------------
// Kernel 1: out[m,:] = x[m,:] @ W + b.  Each block = 4 waves x 16 rows.
// All 8 A-fragments (K=256) are held in registers; the wave then sweeps all
// 32 column blocks -> x is read from HBM exactly ONCE (34 MB, ~1.5us),
// W fragments stream from WGP$/L2 (256 KB resident).
// ---------------------------------------------------------------------------
__global__ __launch_bounds__(128) void xw_gemm(const float* __restrict__ x,
                                               const __bf16* __restrict__ Wpk,
                                               const float* __restrict__ bias,
                                               float* __restrict__ out) {
  const int lane = threadIdx.x & 31;
  const int wave = threadIdx.x >> 5;
  const size_t m0 = (size_t)blockIdx.x * 64 + wave * 16;
  const int r      = lane & 15;
  const int k0base = (lane & 16) ? 8 : 0;
  const float* arow = x + (m0 + r) * (size_t)D_;

  v16bf a[8];
#pragma unroll
  for (int s = 0; s < 8; ++s)
    a[s] = make_afrag_f32(arow + s * 32 + k0base);

  const int colb  = lane & 15;
  const size_t rbase = m0 + ((lane & 16) ? 8 : 0);

  for (int j = 0; j < 32; ++j) {
    v8f acc = {};
#pragma unroll
    for (int s = 0; s < 8; ++s) {
      v16bf b = *(const v16bf*)(Wpk + ((((size_t)j * 8 + s) * 32 + lane) << 4));
      acc = __builtin_amdgcn_wmma_f32_16x16x32_bf16(false, a[s], false, b,
                                                    (short)0, acc, false, false);
    }
    const int col = j * 16 + colb;
    const float bv = bias[col];
#pragma unroll
    for (int rr = 0; rr < 8; ++rr)
      out[(rbase + rr) * (size_t)U_ + col] = acc[rr] + bv;
  }
}

// ---------------------------------------------------------------------------
// Kernel 2: out[:,t,:] = tanh(out[:,t,:] + state_{t-1} @ U), and also emit
// the new state as bf16 into `sout` (ping-pong) so the NEXT step loads its
// A-fragments directly as bf16 (no per-step fp32->bf16 conversion, half the
// state-read bytes).  Grid = 32 col blocks x 4 waves (16 batch rows each).
// ---------------------------------------------------------------------------
__global__ __launch_bounds__(128) void rnn_step(const __bf16* __restrict__ Upk,
                                                const __bf16* __restrict__ sin,
                                                __bf16* __restrict__ sout,
                                                float* __restrict__ out,
                                                int t) {
  const int lane = threadIdx.x & 31;
  const int wave = threadIdx.x >> 5;
  const int jblk = blockIdx.x;                 // 0..31
  const int row0 = wave * 16;                  // batch-row tile base
  const int r      = lane & 15;
  const int k0base = (lane & 16) ? 8 : 0;

  v8f acc = {};
  if (t > 0) {
    const __bf16* arow = sin + (size_t)(row0 + r) * U_;
#pragma unroll 4
    for (int s = 0; s < (U_ >> 5); ++s) {
      v16bf a = make_afrag_bf16(arow + s * 32 + k0base);
      v16bf b = *(const v16bf*)(Upk + ((((size_t)jblk * (U_>>5) + s) * 32 + lane) << 4));
      acc = __builtin_amdgcn_wmma_f32_16x16x32_bf16(false, a, false, b,
                                                    (short)0, acc, false, false);
    }
  }
  const int col = jblk * 16 + (lane & 15);
  const int rb  = row0 + ((lane & 16) ? 8 : 0);
#pragma unroll
  for (int rr = 0; rr < 8; ++rr) {
    const size_t idx = ((size_t)(rb + rr) * T_ + t) * (size_t)U_ + col;
    const float v = tanhf(out[idx] + acc[rr]);
    out[idx] = v;                                            // fp32 result
    sout[(size_t)(rb + rr) * U_ + col] = (__bf16)v;          // bf16 state feed
  }
}

// ---------------------------------------------------------------------------
extern "C" void kernel_launch(void* const* d_in, const int* in_sizes, int n_in,
                              void* d_out, int out_size, void* d_ws, size_t ws_size,
                              hipStream_t stream) {
  (void)in_sizes; (void)n_in; (void)out_size; (void)ws_size;
  const float* x    = (const float*)d_in[0];   // [64,512,256]
  const float* W    = (const float*)d_in[1];   // [256,512]
  const float* Um   = (const float*)d_in[2];   // [512,512]
  const float* bias = (const float*)d_in[3];   // [512]
  float* out = (float*)d_out;                  // [64,512,512]

  __bf16* Wpk = (__bf16*)d_ws;
  __bf16* Upk = (__bf16*)((char*)d_ws + 262144);
  __bf16* st0 = (__bf16*)((char*)d_ws + 786432);
  __bf16* st1 = (__bf16*)((char*)d_ws + 851968);

  // 1) pack weights to bf16 WMMA fragment layout (L2-resident thereafter)
  const int totalPack = D_ * U_ + U_ * U_;
  pack_weights<<<(totalPack + 255) / 256, 256, 0, stream>>>(W, Um, Wpk, Upk);

  // 2) xw = x @ W + b, written straight into d_out (x read exactly once)
  xw_gemm<<<dim3((B_ * T_) / 64), 128, 0, stream>>>(x, Wpk, bias, out);

  // 3) serial recurrence: 512 dependent small GEMM+tanh steps, bf16 state
  //    ping-pongs through the workspace (t=0 ignores sin).
  for (int t = 0; t < T_; ++t) {
    __bf16*       so = (t & 1) ? st1 : st0;
    const __bf16* si = (t & 1) ? st0 : st1;
    rnn_step<<<dim3(U_ / 16), 128, 0, stream>>>(Upk, si, so, out, t);
  }
}
